// ScaledDotProductAttention_17798344475069
// MI455X (gfx1250) — matrix operand order repaired and checked by reading, not compiled, not run on hardware
//
#include <hip/hip_runtime.h>
#include <cmath>

typedef __attribute__((ext_vector_type(16))) _Float16 v16h;
typedef __attribute__((ext_vector_type(8)))  float    v8f;

#define B_   2
#define H_   16
#define S_   2048
#define D_   128
#define QTILE 16
#define NEGINF (-1.0e9f)
#define SCALE_ 0.08838834764831845f   /* 1/sqrt(128) */

// LDS: 16x2048 f32 scores (128KB) + 16 rowmax + 16 rowsum
#define LDS_BYTES ((QTILE * S_ + 32) * sizeof(float))

__global__ __launch_bounds__(32)
void sdpa_fused_wmma_kernel(const float* __restrict__ q,
                            const float* __restrict__ k,
                            const float* __restrict__ v,
                            const int*   __restrict__ mask,
                            float* __restrict__ out,
                            float* __restrict__ attn) {
  extern __shared__ float smem[];
  float* s_scores = smem;                 // [16][2048]
  float* s_rmax   = smem + QTILE * S_;    // [16]
  float* s_rsum   = s_rmax + 16;          // [16]

  const int lane = threadIdx.x;   // 0..31, wave32
  const int hb   = lane >> 4;     // half of wave: 0 or 1
  const int ln   = lane & 15;

  const int tiles_per_bh = S_ / QTILE;            // 128
  const int bh    = blockIdx.x / tiles_per_bh;    // b*H + h
  const int qt    = blockIdx.x % tiles_per_bh;
  const int b     = bh / H_;
  const int qbase = qt * QTILE;

  const float* Q = q + (size_t)bh * S_ * D_;
  const float* K = k + (size_t)bh * S_ * D_;
  const float* V = v + (size_t)bh * S_ * D_;
  const int*   M = mask + (size_t)b * S_ * S_;
  float* O = out + (size_t)bh * S_ * D_;
  float* A = attn + ((size_t)bh * S_ + qbase) * S_;

  // ---------- Q A-fragments: f32 -> f16, 4 K-blocks of 32 ----------
  // A layout (16-bit 16x32): lane row M = ln; half hb=0 -> K {0..15},
  // hb=1 -> K {16..31}, within each 32-wide block.
  v16h qa[4];
  {
    const float* qrow = Q + (size_t)(qbase + ln) * D_;
#pragma unroll
    for (int kb = 0; kb < 4; ++kb) {
#pragma unroll
      for (int i = 0; i < 8; ++i) {
        qa[kb][i]     = (_Float16)qrow[kb * 32 + hb * 16 + i];
        qa[kb][8 + i] = (_Float16)qrow[kb * 32 + hb * 16 + 8 + i];
      }
    }
  }

  // ---------- pass 1: scores = scale * Q K^T + additive mask -> LDS ----------
  for (int nt = 0; nt < tiles_per_bh; ++nt) {
    v8f c = {};
    const float* krow = K + (size_t)(nt * 16 + ln) * D_;  // B col N = key row
#pragma unroll
    for (int kb = 0; kb < 4; ++kb) {
      v16h bfrag;  // B layout (32x16): lane col N = ln; half hb holds K = hb*16 + i
#pragma unroll
      for (int i = 0; i < 16; ++i)
        bfrag[i] = (_Float16)krow[kb * 32 + hb * 16 + i];
      c = __builtin_amdgcn_wmma_f32_16x16x32_f16(false, qa[kb], false, bfrag,
                                                 (short)0, c, false, false);
    }
    const int col = nt * 16 + ln;
#pragma unroll
    for (int r = 0; r < 8; ++r) {
      const int row = r + 8 * hb;  // C layout: VGPR r -> row r + 8*half
      const int mv  = M[(size_t)(qbase + row) * S_ + col];
      s_scores[row * S_ + col] =
          c[r] * SCALE_ + (1.0f - (float)mv) * NEGINF;
    }
  }
  __syncthreads();

  // ---------- row softmax stats (float4-vectorized LDS scans) ----------
  {
    const float4* srow4 =
        reinterpret_cast<const float4*>(s_scores + (size_t)ln * S_);
    float m = -INFINITY;
    for (int i = hb * 256; i < hb * 256 + 256; ++i) {
      float4 t = srow4[i];
      m = fmaxf(m, fmaxf(fmaxf(t.x, t.y), fmaxf(t.z, t.w)));
    }
    m = fmaxf(m, __shfl_xor(m, 16, 32));
    float l = 0.0f;
    for (int i = hb * 256; i < hb * 256 + 256; ++i) {
      float4 t = srow4[i];
      l += __expf(t.x - m) + __expf(t.y - m) +
           __expf(t.z - m) + __expf(t.w - m);
    }
    l += __shfl_xor(l, 16, 32);
    if (hb == 0) { s_rmax[ln] = m; s_rsum[ln] = l; }
  }
  __syncthreads();

  // ---------- normalize in LDS + wide coalesced attn store (b128) ----------
  for (int row = 0; row < QTILE; ++row) {
    const float rm  = s_rmax[row];
    const float inv = 1.0f / s_rsum[row];   // one v_rcp per row, not per element
    float4* srow4 = reinterpret_cast<float4*>(s_scores + (size_t)row * S_);
    float4* arow4 = reinterpret_cast<float4*>(A + (size_t)row * S_);
#pragma unroll 4
    for (int it = 0; it < (S_ / 4) / 32; ++it) {   // 512 float4 per row / 32 lanes
      const int c4 = it * 32 + lane;
      float4 t = srow4[c4];
      t.x = __expf(t.x - rm) * inv;
      t.y = __expf(t.y - rm) * inv;
      t.z = __expf(t.z - rm) * inv;
      t.w = __expf(t.w - rm) * inv;
      srow4[c4] = t;          // keep P for P@V
      arow4[c4] = t;          // the 512MB mandatory store: global_store_b128
    }
  }
  __syncthreads();

  // ---------- pass 2: out = P @ V  (P from LDS, A-layout; V from L2) ----------
  v8f acc[8];
#pragma unroll
  for (int vt = 0; vt < 8; ++vt) acc[vt] = (v8f){};

  for (int kb = 0; kb < S_ / 32; ++kb) {
    v16h afrag;
    const float* prow = s_scores + (size_t)ln * S_ + kb * 32;
#pragma unroll
    for (int i = 0; i < 8; ++i) {
      afrag[i]     = (_Float16)prow[hb * 16 + i];
      afrag[8 + i] = (_Float16)prow[hb * 16 + 8 + i];
    }
    const float* vbase = V + (size_t)(kb * 32 + hb * 16) * D_ + ln;
#pragma unroll
    for (int vt = 0; vt < 8; ++vt) {
      v16h bfrag;  // B[k, n] = V[kb*32 + k, vt*16 + n]
#pragma unroll
      for (int i = 0; i < 16; ++i)
        bfrag[i] = (_Float16)vbase[(size_t)i * D_ + vt * 16];
      acc[vt] = __builtin_amdgcn_wmma_f32_16x16x32_f16(false, afrag, false, bfrag,
                                                       (short)0, acc[vt], false, false);
    }
  }

#pragma unroll
  for (int vt = 0; vt < 8; ++vt) {
#pragma unroll
    for (int r = 0; r < 8; ++r) {
      const int row = r + 8 * hb;
      O[(size_t)(qbase + row) * D_ + vt * 16 + ln] = acc[vt][r];
    }
  }
}

extern "C" void kernel_launch(void* const* d_in, const int* in_sizes, int n_in,
                              void* d_out, int out_size, void* d_ws, size_t ws_size,
                              hipStream_t stream) {
  (void)in_sizes; (void)n_in; (void)out_size; (void)d_ws; (void)ws_size;

  const float* q    = (const float*)d_in[0];
  const float* k    = (const float*)d_in[1];
  const float* v    = (const float*)d_in[2];
  const int*   mask = (const int*)d_in[3];

  float* out  = (float*)d_out;                           // [B,H,S,D] first
  float* attn = out + (size_t)B_ * H_ * S_ * D_;         // then [B,H,S,S]

  // 128KB dynamic LDS > default 64KB cap: raise the attribute (host-side, capture-safe)
  (void)hipFuncSetAttribute((const void*)sdpa_fused_wmma_kernel,
                            hipFuncAttributeMaxDynamicSharedMemorySize,
                            (int)LDS_BYTES);

  const int grid = B_ * H_ * (S_ / QTILE);  // 4096 single-wave workgroups
  sdpa_fused_wmma_kernel<<<dim3(grid), dim3(32), LDS_BYTES, stream>>>(
      q, k, v, mask, out, attn);
}
